// MultiHeadSelfAttention_3298534883514
// MI455X (gfx1250) — matrix operand order repaired and checked by reading, not compiled, not run on hardware
//
#include <hip/hip_runtime.h>
#include <hip/hip_bf16.h>
#include <stdint.h>

#define D_MODEL 1024
#define N_HEADS 16
#define HEAD_DIM 64
#define BATCH 2
#define SEQ 2048

typedef __attribute__((ext_vector_type(16))) __bf16 v16bf;
typedef __attribute__((ext_vector_type(8)))  float  v8f;
typedef __attribute__((ext_vector_type(4)))  int    v4i;
typedef __attribute__((address_space(1))) v4i v4i_as1;   // global int4
typedef __attribute__((address_space(3))) v4i v4i_as3;   // LDS int4

union BFPack { uint4 q[2]; v16bf v; };

#if __has_builtin(__builtin_amdgcn_global_load_async_to_lds_b128)
#define USE_ASYNC_LDS 1
#else
#define USE_ASYNC_LDS 0
#endif

__device__ __forceinline__ unsigned short f2bf(float f) {
    union { float f; unsigned u; } x; x.f = f;
    unsigned u = x.u;
    u += 0x7FFFu + ((u >> 16) & 1u);   // round-to-nearest-even
    return (unsigned short)(u >> 16);
}

__device__ __forceinline__ v8f vzero8() {
    v8f z;
#pragma unroll
    for (int i = 0; i < 8; ++i) z[i] = 0.0f;
    return z;
}

// 16B global -> LDS copy: async on gfx1250 toolchains that expose the builtin,
// synchronous register bounce otherwise.
__device__ __forceinline__ void copy16_g2l(const unsigned short* g, unsigned short* l) {
#if USE_ASYNC_LDS
    __builtin_amdgcn_global_load_async_to_lds_b128(
        (v4i_as1*)g, (v4i_as3*)l, 0, 0);
#else
    *reinterpret_cast<uint4*>(l) = *reinterpret_cast<const uint4*>(g);
#endif
}

__device__ __forceinline__ void wait_async_le2() {
#if USE_ASYNC_LDS
#if __has_builtin(__builtin_amdgcn_s_wait_asynccnt)
    __builtin_amdgcn_s_wait_asynccnt(2);
#else
    asm volatile("s_wait_asynccnt 0x2" ::: "memory");
#endif
#endif
}

__device__ __forceinline__ void wait_async_le0() {
#if USE_ASYNC_LDS
#if __has_builtin(__builtin_amdgcn_s_wait_asynccnt)
    __builtin_amdgcn_s_wait_asynccnt(0);
#else
    asm volatile("s_wait_asynccnt 0x0" ::: "memory");
#endif
#endif
}

// A tile 16(M) x 32(K), bf16, row-major source (K contiguous), leading dim ld.
// Lane<16: row M=lane, v[0..15]=K0..15.
// Lane>=16: row M=lane-16, v[0..15]=K16..31.
__device__ __forceinline__ v16bf load_a_tile(const unsigned short* src, int ld, int lane) {
    int r  = lane & 15;
    int kb = (lane >> 4) << 4;               // 0 or 16
    BFPack u;
    u.q[0] = *reinterpret_cast<const uint4*>(src + (size_t)r * ld + kb);
    u.q[1] = *reinterpret_cast<const uint4*>(src + (size_t)r * ld + kb + 8);
    return u.v;
}

// B tile 32(K) x 16(N), bf16. B[k,n] = src[n, k] (src row-major, row n, K contiguous).
// Lane<16: col N=lane, v[0..15]=K0..15. Lane>=16: col N=lane-16, v[0..15]=K16..31.
__device__ __forceinline__ v16bf load_b_tile(const unsigned short* src, size_t ld, int lane) {
    int r   = lane & 15;
    int off = (lane >> 4) << 4;              // 0 or 16
    const uint4* p = reinterpret_cast<const uint4*>(src + (size_t)r * ld + off);
    BFPack u;
    u.q[0] = p[0];
    u.q[1] = p[1];
    return u.v;
}

__device__ __forceinline__ v8f wmma_bf16(v16bf a, v16bf b, v8f c) {
    return __builtin_amdgcn_wmma_f32_16x16x32_bf16(false, a, false, b, (short)0, c, false, false);
}

__global__ void cvt_f32_bf16(const float* __restrict__ in, unsigned short* __restrict__ out, int n) {
    int i = blockIdx.x * blockDim.x + threadIdx.x;
    if (i < n) out[i] = f2bf(in[i]);
}

__global__ void zero_f32(float* __restrict__ p, int n) {
    int i = blockIdx.x * blockDim.x + threadIdx.x;
    if (i < n) p[i] = 0.0f;
}

// C = A(MxK) * B(NxK)^T + bias  with bf16 inputs, fp32 accum.
// Block = 4 waves, computes a 256(M) x 64(N) tile. Wave w owns rows [tileM+64w, +64):
// 4 A-tiles x 4 B-tiles = 16 WMMAs per 32-wide k-step. The shared 64x32 B panel is
// staged in LDS (double-buffered) via async global->LDS copies.
// mode 0: store bf16 as Q/K layout [B,H,S,64]
// mode 1: store bf16 as V^T layout [B,H,64,S]
// mode 2: store fp32 row-major [M,N]
__global__ __launch_bounds__(128) void gemm_bf16(
    const unsigned short* __restrict__ A,
    const unsigned short* __restrict__ Bm,
    const float* __restrict__ bias,
    void* __restrict__ out,
    int M, int N, int K, int mode)
{
    __shared__ __align__(16) unsigned short bpan[2][64][32];

    const int lane = threadIdx.x & 31;
    const int w    = threadIdx.x >> 5;
    const int tid  = threadIdx.x;
    const int nTN  = N >> 6;
    const int tileM = (blockIdx.x / nTN) << 8;     // 256-row tiles
    const int tileN = (blockIdx.x % nTN) << 6;
    const int m0 = tileM + (w << 6);               // wave's 64 rows

    v8f acc[4][4];
#pragma unroll
    for (int at = 0; at < 4; ++at)
#pragma unroll
        for (int nt = 0; nt < 4; ++nt) acc[at][nt] = vzero8();

    // Panel fill: 64 rows x 64B = 4KB = 256 x 16B chunks; 128 threads x 2 chunks.
    auto issue_panel = [&](int buf, int kk) {
#pragma unroll
        for (int j = 0; j < 2; ++j) {
            int c   = tid * 2 + j;
            int row = c >> 2;
            int seg = (c & 3) << 3;                // element offset (8 bf16 = 16B)
            copy16_g2l(Bm + (size_t)(tileN + row) * K + kk + seg,
                       &bpan[buf][row][seg]);
        }
    };

    const int ksteps = K >> 5;
    issue_panel(0, 0);
    for (int ks = 0; ks < ksteps; ++ks) {
        const int kk  = ks << 5;
        const int cur = ks & 1;
        const bool more = (ks + 1) < ksteps;
        if (more) issue_panel(cur ^ 1, kk + 32);
        if (more) wait_async_le2(); else wait_async_le0();
        __syncthreads();                           // panel[cur] visible to all waves

        v16bf a[4];
#pragma unroll
        for (int at = 0; at < 4; ++at) {
            if (more)
                __builtin_prefetch(A + (size_t)(m0 + at * 16) * K + kk + 32, 0, 1);
            a[at] = load_a_tile(A + (size_t)(m0 + at * 16) * K + kk, K, lane);
        }
#pragma unroll
        for (int nt = 0; nt < 4; ++nt) {
            v16bf bt = load_b_tile(&bpan[cur][nt * 16][0], 32, lane);
#pragma unroll
            for (int at = 0; at < 4; ++at)
                acc[at][nt] = wmma_bf16(a[at], bt, acc[at][nt]);
        }
        __syncthreads();                           // all reads of panel[cur] done
    }

    const int hi8 = (lane >> 4) << 3;
#pragma unroll
    for (int at = 0; at < 4; ++at) {
#pragma unroll
        for (int nt = 0; nt < 4; ++nt) {
            const int e  = tileN + nt * 16 + (lane & 15);
            const float bv = bias ? bias[e] : 0.0f;
#pragma unroll
            for (int i = 0; i < 8; ++i) {
                float val = acc[at][nt][i] + bv;
                int mrow = m0 + at * 16 + i + hi8;
                if (mode == 2) {
                    ((float*)out)[(size_t)mrow * N + e] = val;
                } else {
                    int bb = mrow >> 11;            // / SEQ
                    int s  = mrow & (SEQ - 1);
                    int h  = e >> 6;
                    int d  = e & (HEAD_DIM - 1);
                    if (mode == 0) {
                        ((unsigned short*)out)[(((size_t)(bb * N_HEADS + h) * SEQ + s) << 6) + d] = f2bf(val);
                    } else {
                        ((unsigned short*)out)[((size_t)(bb * N_HEADS + h) * HEAD_DIM + d) * SEQ + s] = f2bf(val);
                    }
                }
            }
        }
    }
}

// Attention: block = (b, h, 64 query rows); 4 waves, each wave 16 query rows.
// Two passes: (A) online softmax stats, (B) probs -> attn_mean (+atomic) and ctx via WMMA.
__global__ __launch_bounds__(128) void attn_kernel(
    const unsigned short* __restrict__ Q,
    const unsigned short* __restrict__ K,
    const unsigned short* __restrict__ Vt,
    const int* __restrict__ mask,
    unsigned short* __restrict__ ctx,
    float* __restrict__ attn_mean)
{
    __shared__ __align__(16) unsigned short lds_p[4][16 * 32];

    const int lane = threadIdx.x & 31;
    const int w    = threadIdx.x >> 5;
    const int qblk = blockIdx.x & 31;        // S/64 = 32
    const int h    = (blockIdx.x >> 5) & (N_HEADS - 1);
    const int b    = blockIdx.x >> 9;
    const int q0   = qblk * 64 + w * 16;
    const int hi8  = (lane >> 4) << 3;

    const unsigned short* Qbh = Q  + (size_t)(b * N_HEADS + h) * SEQ * HEAD_DIM;
    const unsigned short* Kbh = K  + (size_t)(b * N_HEADS + h) * SEQ * HEAD_DIM;
    const unsigned short* Vtb = Vt + (size_t)(b * N_HEADS + h) * HEAD_DIM * SEQ;

    v16bf a0 = load_a_tile(Qbh + (size_t)q0 * HEAD_DIM,      HEAD_DIM, lane);
    v16bf a1 = load_a_tile(Qbh + (size_t)q0 * HEAD_DIM + 32, HEAD_DIM, lane);

    float mrow[8], lrow[8];
#pragma unroll
    for (int i = 0; i < 8; ++i) { mrow[i] = -3.0e38f; lrow[i] = 0.0f; }

    // ---- Pass A: online per-lane (m, l) over all key tiles ----
    for (int kt = 0; kt < SEQ / 16; ++kt) {
        const unsigned short* kbase = Kbh + (size_t)kt * 16 * HEAD_DIM;
        v16bf b0 = load_b_tile(kbase,      HEAD_DIM, lane);
        v16bf b1 = load_b_tile(kbase + 32, HEAD_DIM, lane);
        v8f s = wmma_bf16(a0, b0, vzero8());
        s = wmma_bf16(a1, b1, s);
        int mv = mask[b * SEQ + kt * 16 + (lane & 15)];
#pragma unroll
        for (int i = 0; i < 8; ++i) {
            float sc = mv ? s[i] * 0.125f : -1.0e9f;
            float nm = fmaxf(mrow[i], sc);
            lrow[i] = lrow[i] * __expf(mrow[i] - nm) + __expf(sc - nm);
            mrow[i] = nm;
        }
    }
    // reduce (m,l) across the 16-lane half (row lives in one VGPR slot across 16 lanes)
#pragma unroll
    for (int i = 0; i < 8; ++i) {
#pragma unroll
        for (int off = 1; off < 16; off <<= 1) {
            float mo = __shfl_xor(mrow[i], off, 32);
            float lo = __shfl_xor(lrow[i], off, 32);
            float nm = fmaxf(mrow[i], mo);
            lrow[i] = lrow[i] * __expf(mrow[i] - nm) + lo * __expf(mo - nm);
            mrow[i] = nm;
        }
    }
    float linv[8];
#pragma unroll
    for (int i = 0; i < 8; ++i) linv[i] = (lrow[i] > 0.0f) ? 1.0f / lrow[i] : 0.0f;

    // ---- Pass B: probs -> attn_mean + ctx ----
    v8f c[4];
#pragma unroll
    for (int nt = 0; nt < 4; ++nt) c[nt] = vzero8();

    for (int kc = 0; kc < SEQ / 32; ++kc) {
#pragma unroll
        for (int sub = 0; sub < 2; ++sub) {
            int kt = kc * 2 + sub;
            const unsigned short* kbase = Kbh + (size_t)kt * 16 * HEAD_DIM;
            v16bf b0 = load_b_tile(kbase,      HEAD_DIM, lane);
            v16bf b1 = load_b_tile(kbase + 32, HEAD_DIM, lane);
            v8f s = wmma_bf16(a0, b0, vzero8());
            s = wmma_bf16(a1, b1, s);
            int kcol = kt * 16 + (lane & 15);
            int mv = mask[b * SEQ + kcol];
#pragma unroll
            for (int i = 0; i < 8; ++i) {
                float pv = mv ? __expf(s[i] * 0.125f - mrow[i]) * linv[i] : 0.0f;
                atomicAdd(&attn_mean[((size_t)b * SEQ + (q0 + i + hi8)) * SEQ + kcol],
                          pv * (1.0f / N_HEADS));
                lds_p[w][(i + hi8) * 32 + sub * 16 + (lane & 15)] = f2bf(pv);
            }
        }
        __syncthreads();    // uniform across block; orders LDS write -> cross-lane read
        v16bf pa = load_a_tile(lds_p[w], 32, lane);
#pragma unroll
        for (int nt = 0; nt < 4; ++nt) {
            v16bf vb = load_b_tile(Vtb + (size_t)(nt * 16) * SEQ + kc * 32, (size_t)SEQ, lane);
            c[nt] = wmma_bf16(pa, vb, c[nt]);
        }
    }

    // store ctx (bf16) into [B*S, D_MODEL] layout for the output projection
#pragma unroll
    for (int nt = 0; nt < 4; ++nt) {
        int d = nt * 16 + (lane & 15);
#pragma unroll
        for (int i = 0; i < 8; ++i) {
            int qrow = q0 + i + hi8;
            ctx[(size_t)(b * SEQ + qrow) * D_MODEL + h * HEAD_DIM + d] = f2bf(c[nt][i]);
        }
    }
}

extern "C" void kernel_launch(void* const* d_in, const int* in_sizes, int n_in,
                              void* d_out, int out_size, void* d_ws, size_t ws_size,
                              hipStream_t stream)
{
    (void)in_sizes; (void)n_in; (void)out_size; (void)ws_size;

    const float* x    = (const float*)d_in[0];
    const int*   mask = (const int*)  d_in[1];
    const float* Wq   = (const float*)d_in[2];
    const float* bq   = (const float*)d_in[3];
    const float* Wk   = (const float*)d_in[4];
    const float* bk   = (const float*)d_in[5];
    const float* Wv   = (const float*)d_in[6];
    const float* bv   = (const float*)d_in[7];
    const float* Wo   = (const float*)d_in[8];
    const float* bo   = (const float*)d_in[9];

    float* out       = (float*)d_out;
    float* attn_mean = out + (size_t)BATCH * SEQ * D_MODEL;

    const int nBS  = BATCH * SEQ;            // 4096
    const int nXD  = nBS * D_MODEL;          // 4,194,304
    const int nWW  = D_MODEL * D_MODEL;      // 1,048,576

    unsigned short* p    = (unsigned short*)d_ws;
    unsigned short* xbf  = p; p += (size_t)nXD;
    unsigned short* wqb  = p; p += (size_t)nWW;
    unsigned short* wkb  = p; p += (size_t)nWW;
    unsigned short* wvb  = p; p += (size_t)nWW;
    unsigned short* wob  = p; p += (size_t)nWW;
    unsigned short* Qb   = p; p += (size_t)nXD;
    unsigned short* Kb   = p; p += (size_t)nXD;
    unsigned short* Vtb  = p; p += (size_t)nXD;
    unsigned short* ctxb = p; p += (size_t)nXD;

    // zero attn-mean region (accumulated with atomics)
    {
        int n = BATCH * SEQ * SEQ;
        zero_f32<<<(n + 255) / 256, 256, 0, stream>>>(attn_mean, n);
    }

    cvt_f32_bf16<<<(nXD + 255) / 256, 256, 0, stream>>>(x,  xbf, nXD);
    cvt_f32_bf16<<<(nWW + 255) / 256, 256, 0, stream>>>(Wq, wqb, nWW);
    cvt_f32_bf16<<<(nWW + 255) / 256, 256, 0, stream>>>(Wk, wkb, nWW);
    cvt_f32_bf16<<<(nWW + 255) / 256, 256, 0, stream>>>(Wv, wvb, nWW);
    cvt_f32_bf16<<<(nWW + 255) / 256, 256, 0, stream>>>(Wo, wob, nWW);

    const int gemmGrid = (nBS / 256) * (D_MODEL / 64);   // 16 * 16 = 256
    gemm_bf16<<<gemmGrid, 128, 0, stream>>>(xbf, wqb, bq, Qb,  nBS, D_MODEL, D_MODEL, 0);
    gemm_bf16<<<gemmGrid, 128, 0, stream>>>(xbf, wkb, bk, Kb,  nBS, D_MODEL, D_MODEL, 0);
    gemm_bf16<<<gemmGrid, 128, 0, stream>>>(xbf, wvb, bv, Vtb, nBS, D_MODEL, D_MODEL, 1);

    attn_kernel<<<BATCH * N_HEADS * (SEQ / 64), 128, 0, stream>>>(Qb, Kb, Vtb, mask, ctxb, attn_mean);

    gemm_bf16<<<gemmGrid, 128, 0, stream>>>(ctxb, wob, bo, out, nBS, D_MODEL, D_MODEL, 2);
}
